// SimpleNetMax_2542620639563
// MI455X (gfx1250) — matrix operand order repaired and checked by reading, not compiled, not run on hardware
//
#include <hip/hip_runtime.h>

typedef __attribute__((ext_vector_type(16))) __bf16        v16bf;
typedef __attribute__((ext_vector_type(8)))  float         v8f;
typedef __attribute__((ext_vector_type(8)))  unsigned int  v8u;

#define DIN 128
#define DH  64

static __device__ __forceinline__ unsigned int pack_bf16_pair(float lo, float hi) {
    unsigned short a = __builtin_bit_cast(unsigned short, (__bf16)lo);
    unsigned short b = __builtin_bit_cast(unsigned short, (__bf16)hi);
    return (unsigned int)a | ((unsigned int)b << 16);
}

// ---------------------------------------------------------------------------
// Degree / normalization
// ---------------------------------------------------------------------------
__global__ __launch_bounds__(256) void deg_init_kernel(float* deg, int n) {
    int i = blockIdx.x * 256 + threadIdx.x;
    if (i < n) deg[i] = 1.0f;                      // self loop contributes 1
}

__global__ __launch_bounds__(256) void deg_edge_kernel(const long long* __restrict__ ei,
                                                       float* __restrict__ deg, int nedges) {
    long long e = (long long)blockIdx.x * 256 + threadIdx.x;
    long long stride = (long long)gridDim.x * 256;
    for (; e < nedges; e += stride) {
        int r = (int)ei[e];                        // row = edge_index[0]
        atomicAdd(&deg[r], 1.0f);
    }
}

__global__ __launch_bounds__(256) void dis_kernel(float* deg, int n) {
    int i = blockIdx.x * 256 + threadIdx.x;
    if (i < n) deg[i] = rsqrtf(deg[i]);            // deg >= 1 always (self loop)
}

// ---------------------------------------------------------------------------
// xw = x @ W1 via v_wmma_f32_16x16x32_bf16. Block = 8 waves, 16 rows each.
// W1 is pre-swizzled into LDS in fragment order so each lane's B fragment is a
// contiguous 32B block (2x ds_load_b128 instead of 16x ds_load_u16).
// ---------------------------------------------------------------------------
__global__ __launch_bounds__(256) void xw_gemm_kernel(const float* __restrict__ x,
                                                      const float* __restrict__ W1,
                                                      float* __restrict__ xw, int n) {
    // sB[((ks*2+half)*64 + n)*8 + j] = packed bf16 pair (W1[k][n], W1[k+1][n]),
    // k = ks*32 + half*16 + 2j  (contiguous k order 16*half + s, same as the A fragment)
    __shared__ __align__(32) unsigned int sB[4096];     // 16 KB
    int tid = threadIdx.x;
    for (int idx = tid; idx < 4096; idx += 256) {
        int j    = idx & 7;
        int nc   = (idx >> 3) & 63;
        int half = (idx >> 9) & 1;
        int ks   = idx >> 10;
        int k    = ks * 32 + half * 16 + 2 * j;
        sB[idx]  = pack_bf16_pair(W1[k * DH + nc], W1[(k + 1) * DH + nc]);
    }
    __syncthreads();

    int wave = tid >> 5;
    int lane = tid & 31;
    int m0   = blockIdx.x * 128 + wave * 16;
    int m    = lane & 15;
    int rowA = m0 + m;
    if (rowA >= n) rowA = n - 1;                   // clamp: keeps EXEC all-1 for WMMA
    const float* xrow = x + (long long)rowA * DIN;

    const int abase = (lane < 16) ? 0 : 16;        // A layout: K-pair base per half-wave
    const int half  = (lane < 16) ? 0 : 1;         // B layout: K base per half-wave
    const int ncol0 = lane & 15;

    v8f acc[4];
#pragma unroll
    for (int t = 0; t < 4; ++t) { v8f z = {}; acc[t] = z; }

#pragma unroll
    for (int ks = 0; ks < 4; ++ks) {               // K = 128 in 4 steps of 32
        int kk = ks * 32;
        v8u au;
#pragma unroll
        for (int j = 0; j < 8; ++j) {
            int k = kk + abase + 2 * j;
            float2 f = *(const float2*)(xrow + k);
            au[j] = pack_bf16_pair(f.x, f.y);      // -> v_cvt_pk_bf16_f32
        }
        v16bf A = __builtin_bit_cast(v16bf, au);

        const unsigned int* sBk = &sB[(unsigned)((ks * 2 + half) * 64) * 8u];
#pragma unroll
        for (int t = 0; t < 4; ++t) {
            v16bf B = *(const v16bf*)&sBk[(t * 16 + ncol0) * 8]; // 2x ds_load_b128
            acc[t] = __builtin_amdgcn_wmma_f32_16x16x32_bf16(
                false, A, false, B, (short)0, acc[t], false, false);
        }
    }

    int mro = (lane < 16) ? 0 : 8;                 // D layout: vgpr r -> M = r + 8*(hi half)
#pragma unroll
    for (int t = 0; t < 4; ++t) {
#pragma unroll
        for (int r = 0; r < 8; ++r) {
            int mr = m0 + mro + r;
            if (mr < n) xw[(long long)mr * DH + t * 16 + ncol0] = acc[t][r];
        }
    }
}

// ---------------------------------------------------------------------------
// Edge aggregation, 64-wide messages: agg[row] += xw[col] * dis[row]*dis[col]
// 16 lanes per edge, float4 per lane, f32 near-atomics in L2.
// ---------------------------------------------------------------------------
__global__ __launch_bounds__(256) void agg64_kernel(const long long* __restrict__ ei,
                                                    const float* __restrict__ xw,
                                                    const float* __restrict__ dis,
                                                    float* __restrict__ agg, int nedges) {
    long long gid = (long long)blockIdx.x * 256 + threadIdx.x;
    int sub = (int)(gid & 15);
    long long e = gid >> 4;
    long long stride = ((long long)gridDim.x * 256) >> 4;
    for (; e < nedges; e += stride) {
        int r = (int)ei[e];
        int c = (int)ei[nedges + e];
        float nrm = dis[r] * dis[c];
        float4 v = ((const float4*)(xw + (long long)c * DH))[sub];
        float* dst = agg + (long long)r * DH + sub * 4;
        atomicAdd(dst + 0, v.x * nrm);
        atomicAdd(dst + 1, v.y * nrm);
        atomicAdd(dst + 2, v.z * nrm);
        atomicAdd(dst + 3, v.w * nrm);
    }
}

// ---------------------------------------------------------------------------
// Fused: h = relu(agg + xw*dis^2 + b1) (in place over agg),
//        hw = h @ W2 (wave reduction), running column max of h (bit-max, h>=0)
// One wave per node, grid-stride.
// ---------------------------------------------------------------------------
__global__ __launch_bounds__(256) void finalize_h_kernel(float* __restrict__ agg,
                                                         const float* __restrict__ xw,
                                                         const float* __restrict__ dis,
                                                         const float* __restrict__ b1,
                                                         const float* __restrict__ W2,
                                                         float* __restrict__ hw,
                                                         unsigned int* __restrict__ xmax_bits,
                                                         int n) {
    __shared__ unsigned int smax[DH];
    int tid = threadIdx.x, lane = tid & 31, wv = tid >> 5;
    if (tid < DH) smax[tid] = 0u;
    __syncthreads();

    float w2a = W2[lane], w2b = W2[lane + 32];
    float b1a = b1[lane], b1b = b1[lane + 32];
    unsigned int mx0 = 0u, mx1 = 0u;

    int i = blockIdx.x * 8 + wv;
    int istep = gridDim.x * 8;
    for (; i < n; i += istep) {
        float d2 = dis[i]; d2 *= d2;
        long long base = (long long)i * DH;
        float h0 = agg[base + lane]      + xw[base + lane]      * d2 + b1a;
        float h1 = agg[base + lane + 32] + xw[base + lane + 32] * d2 + b1b;
        h0 = fmaxf(h0, 0.0f);
        h1 = fmaxf(h1, 0.0f);
        agg[base + lane]      = h0;
        agg[base + lane + 32] = h1;
        mx0 = max(mx0, __float_as_uint(h0));       // valid: relu output >= 0
        mx1 = max(mx1, __float_as_uint(h1));
        float p = h0 * w2a + h1 * w2b;
        for (int off = 16; off > 0; off >>= 1) p += __shfl_down(p, off, 32);
        if (lane == 0) hw[i] = p;
    }
    atomicMax(&smax[lane], mx0);
    atomicMax(&smax[lane + 32], mx1);
    __syncthreads();
    if (tid < DH) atomicMax(&xmax_bits[tid], smax[tid]);
}

// ---------------------------------------------------------------------------
// Edge aggregation, scalar messages: agg2[row] += hw[col]*dis[row]*dis[col]
// ---------------------------------------------------------------------------
__global__ __launch_bounds__(256) void agg1_kernel(const long long* __restrict__ ei,
                                                   const float* __restrict__ hw,
                                                   const float* __restrict__ dis,
                                                   float* __restrict__ agg2, int nedges) {
    long long e = (long long)blockIdx.x * 256 + threadIdx.x;
    long long stride = (long long)gridDim.x * 256;
    for (; e < nedges; e += stride) {
        int r = (int)ei[e];
        int c = (int)ei[nedges + e];
        atomicAdd(&agg2[r], hw[c] * dis[r] * dis[c]);
    }
}

// ---------------------------------------------------------------------------
// probs_all = agg2 + hw*dis^2 + b2; per-block ready count (wave32 ballot)
// ---------------------------------------------------------------------------
__global__ __launch_bounds__(256) void probs_count_kernel(const float* __restrict__ agg2,
                                                          const float* __restrict__ hw,
                                                          const float* __restrict__ dis,
                                                          const float* __restrict__ b2,
                                                          const int* __restrict__ ready,
                                                          float* __restrict__ probs_all,
                                                          int* __restrict__ blockCounts, int n) {
    __shared__ int cnt;
    if (threadIdx.x == 0) cnt = 0;
    __syncthreads();
    int i = blockIdx.x * 256 + threadIdx.x;
    int rdy = 0;
    if (i < n) {
        float d = dis[i];
        probs_all[i] = agg2[i] + hw[i] * d * d + b2[0];
        rdy = (ready[i] != 0) ? 1 : 0;
    }
    unsigned long long bal = __ballot(rdy);
    int wcount = __popcll(bal);
    if ((threadIdx.x & 31) == 0) atomicAdd(&cnt, wcount);
    __syncthreads();
    if (threadIdx.x == 0) blockCounts[blockIdx.x] = cnt;
}

__global__ void scan_kernel(const int* __restrict__ blockCounts,
                            int* __restrict__ blockOffsets, int nblocks,
                            int* __restrict__ totalK) {
    int s = 0;                                      // ~400 blocks: trivial serial scan
    for (int b = 0; b < nblocks; ++b) { blockOffsets[b] = s; s += blockCounts[b]; }
    *totalK = s;
}

__global__ __launch_bounds__(256) void scatter_kernel(const float* __restrict__ probs_all,
                                                      const int* __restrict__ ready,
                                                      const int* __restrict__ blockOffsets,
                                                      float* __restrict__ logits, int n) {
    __shared__ int wsum[8];
    int i = blockIdx.x * 256 + threadIdx.x;
    int lane = threadIdx.x & 31, wv = threadIdx.x >> 5;
    int rdy = (i < n) && (ready[i] != 0);
    unsigned int bal = (unsigned int)__ballot(rdy);          // wave32: low 32 bits
    int wpre = __popc(bal & ((1u << lane) - 1u));
    if (lane == 0) wsum[wv] = __popc(bal);
    __syncthreads();
    int woff = 0;
    for (int w = 0; w < wv; ++w) woff += wsum[w];
    if (rdy) logits[blockOffsets[blockIdx.x] + woff + wpre] = probs_all[i];
}

// ---------------------------------------------------------------------------
// Head: prob_nothing = xmax.Wdn + bdn -> logits[K];  v = xmax.Wv + bv
// ---------------------------------------------------------------------------
__global__ __launch_bounds__(64) void head_kernel(const unsigned int* __restrict__ xmax_bits,
                                                  const float* __restrict__ Wdn,
                                                  const float* __restrict__ bdn,
                                                  const float* __restrict__ Wv,
                                                  const float* __restrict__ bv,
                                                  const int* __restrict__ totalK,
                                                  float* __restrict__ logits,
                                                  float* __restrict__ vout) {
    __shared__ float sdn[DH], sv[DH];
    int t = threadIdx.x;
    float xm = __uint_as_float(xmax_bits[t]);
    sdn[t] = xm * Wdn[t];
    sv[t]  = xm * Wv[t];
    __syncthreads();
    for (int off = 32; off > 0; off >>= 1) {
        if (t < off) { sdn[t] += sdn[t + off]; sv[t] += sv[t + off]; }
        __syncthreads();
    }
    if (t == 0) { logits[*totalK] = sdn[0] + bdn[0]; vout[0] = sv[0] + bv[0]; }
}

// ---------------------------------------------------------------------------
// Softmax over K+1 logits -> out[0..K]; out[K+1] = v
// ---------------------------------------------------------------------------
__global__ __launch_bounds__(1024) void softmax_kernel(const float* __restrict__ logits,
                                                       const int* __restrict__ totalK,
                                                       const float* __restrict__ vout,
                                                       float* __restrict__ out) {
    __shared__ float red[32];
    int M = *totalK + 1;
    int t = threadIdx.x, lane = t & 31, wv = t >> 5;

    float m = -3.0e38f;
    for (int i = t; i < M; i += 1024) m = fmaxf(m, logits[i]);
    for (int off = 16; off > 0; off >>= 1) m = fmaxf(m, __shfl_down(m, off, 32));
    if (lane == 0) red[wv] = m;
    __syncthreads();
    if (t < 32) {
        float mm = red[t];
        for (int off = 16; off > 0; off >>= 1) mm = fmaxf(mm, __shfl_down(mm, off, 32));
        if (t == 0) red[0] = mm;
    }
    __syncthreads();
    m = red[0];
    __syncthreads();

    float s = 0.0f;
    for (int i = t; i < M; i += 1024) s += __expf(logits[i] - m);
    for (int off = 16; off > 0; off >>= 1) s += __shfl_down(s, off, 32);
    if (lane == 0) red[wv] = s;
    __syncthreads();
    if (t < 32) {
        float ss = red[t];
        for (int off = 16; off > 0; off >>= 1) ss += __shfl_down(ss, off, 32);
        if (t == 0) red[0] = ss;
    }
    __syncthreads();
    float inv = 1.0f / red[0];

    for (int i = t; i < M; i += 1024) out[i] = __expf(logits[i] - m) * inv;
    if (t == 0) out[M] = vout[0];
}

// ---------------------------------------------------------------------------
extern "C" void kernel_launch(void* const* d_in, const int* in_sizes, int n_in,
                              void* d_out, int out_size, void* d_ws, size_t ws_size,
                              hipStream_t stream) {
    const float*     x     = (const float*)d_in[0];
    const long long* ei    = (const long long*)d_in[1];
    const int*       ready = (const int*)d_in[2];
    const float*     W1    = (const float*)d_in[3];
    const float*     b1    = (const float*)d_in[4];
    const float*     W2    = (const float*)d_in[5];
    const float*     b2    = (const float*)d_in[6];
    const float*     Wdn   = (const float*)d_in[7];
    const float*     bdn   = (const float*)d_in[8];
    const float*     Wv    = (const float*)d_in[9];
    const float*     bv    = (const float*)d_in[10];

    const int n      = in_sizes[0] / DIN;
    const int nedges = in_sizes[1] / 2;
    const int nB     = (n + 255) / 256;

    char* p = (char*)d_ws;
    auto alloc = [&](size_t bytes) -> char* {
        char* q = p; p += (bytes + 255) & ~(size_t)255; return q;
    };
    float*        xw        = (float*)alloc((size_t)n * DH * 4);
    float*        agg       = (float*)alloc((size_t)n * DH * 4);   // becomes h in place
    float*        dis       = (float*)alloc((size_t)n * 4);        // deg -> rsqrt(deg)
    float*        hw        = (float*)alloc((size_t)n * 4);
    float*        agg2      = (float*)alloc((size_t)n * 4);
    float*        probs_all = (float*)alloc((size_t)n * 4);
    float*        logits    = (float*)alloc((size_t)(n + 1) * 4);
    unsigned int* xmax_bits = (unsigned int*)alloc(DH * 4);
    int*          blkCnt    = (int*)alloc((size_t)nB * 4);
    int*          blkOff    = (int*)alloc((size_t)nB * 4);
    int*          totalK    = (int*)alloc(4);
    float*        vout      = (float*)alloc(4);

    // per-call scratch init (capture-legal memset nodes)
    hipMemsetAsync(agg,       0, (size_t)n * DH * 4, stream);
    hipMemsetAsync(agg2,      0, (size_t)n * 4,      stream);
    hipMemsetAsync(xmax_bits, 0, DH * 4,             stream);

    // degrees / symmetric norm
    deg_init_kernel<<<nB, 256, 0, stream>>>(dis, n);
    deg_edge_kernel<<<(nedges + 255) / 256, 256, 0, stream>>>(ei, dis, nedges);
    dis_kernel<<<nB, 256, 0, stream>>>(dis, n);

    // dense GEMM via WMMA bf16
    xw_gemm_kernel<<<(n + 127) / 128, 256, 0, stream>>>(x, W1, xw, n);

    // conv1 aggregation (64-wide messages)
    agg64_kernel<<<(int)(((long long)nedges * 16 + 255) / 256), 256, 0, stream>>>(
        ei, xw, dis, agg, nedges);

    // fused bias+relu + h@W2 + column max
    finalize_h_kernel<<<1024, 256, 0, stream>>>(agg, xw, dis, b1, W2, hw, xmax_bits, n);

    // conv2 aggregation (scalar messages)
    agg1_kernel<<<(nedges + 255) / 256, 256, 0, stream>>>(ei, hw, dis, agg2, nedges);

    // probs_all + ready compaction
    probs_count_kernel<<<nB, 256, 0, stream>>>(agg2, hw, dis, b2, ready, probs_all, blkCnt, n);
    scan_kernel<<<1, 1, 0, stream>>>(blkCnt, blkOff, nB, totalK);
    scatter_kernel<<<nB, 256, 0, stream>>>(probs_all, ready, blkOff, logits, n);

    // heads + softmax + output
    head_kernel<<<1, 64, 0, stream>>>(xmax_bits, Wdn, bdn, Wv, bv, totalK, logits, vout);
    softmax_kernel<<<1, 1024, 0, stream>>>(logits, totalK, vout, (float*)d_out);
}
